// GCNModel_83494164234356
// MI455X (gfx1250) — matrix operand order repaired and checked by reading, not compiled, not run on hardware
//
#include <hip/hip_runtime.h>
#include <math.h>

#define N_NODES  50000
#define N_EDGES  800000
#define N_GRAPHS 128
#define IN_FEATS 128
#define HIDDEN   256

typedef __attribute__((ext_vector_type(16))) __bf16 v16bf;
typedef __attribute__((ext_vector_type(8)))  float  v8f;

union FragBF { v16bf v; uint4 u[2]; };

__device__ __forceinline__ unsigned short f32_to_bf16(float f) {
  unsigned u = __float_as_uint(f);
  u += 0x7FFFu + ((u >> 16) & 1u);   // round-to-nearest-even
  return (unsigned short)(u >> 16);
}

// ---------------------------------------------------------------- utilities
__global__ void k_zero(float* __restrict__ p, int n) {
  int i = blockIdx.x * blockDim.x + threadIdx.x;
  int stride = gridDim.x * blockDim.x;
  for (; i < n; i += stride) p[i] = 0.0f;
}

__global__ void k_deg(const int* __restrict__ src, const int* __restrict__ dst,
                      float* __restrict__ outdeg, float* __restrict__ indeg) {
  int e = blockIdx.x * blockDim.x + threadIdx.x;
  if (e >= N_EDGES) return;
  atomicAdd(&outdeg[src[e]], 1.0f);
  atomicAdd(&indeg[dst[e]], 1.0f);
}

__global__ void k_counts(const int* __restrict__ gid, float* __restrict__ counts) {
  int i = blockIdx.x * blockDim.x + threadIdx.x;
  if (i >= N_NODES) return;
  atomicAdd(&counts[gid[i]], 1.0f);
}

__global__ void k_rsqrt(float* __restrict__ d, int n) {
  int i = blockIdx.x * blockDim.x + threadIdx.x;
  if (i < n) d[i] = rsqrtf(fmaxf(d[i], 1.0f));
}

// W [K][N] f32  ->  Wt [N][K] bf16   (so B fragments are contiguous per lane)
__global__ void k_wt_bf16(const float* __restrict__ W, unsigned short* __restrict__ Wt,
                          int K, int N) {
  int i = blockIdx.x * blockDim.x + threadIdx.x;
  if (i >= K * N) return;
  int k = i / N, n = i - k * N;
  Wt[n * K + k] = f32_to_bf16(W[i]);
}

// ---------------------------------------------------------------- SpMM scatter
// agg[dst] += h[src] * out_norm[src], 4 floats per thread. shift4 = log2(F/4).
__global__ void k_scatter(const float* __restrict__ h, const int* __restrict__ src,
                          const int* __restrict__ dst, const float* __restrict__ onorm,
                          float* __restrict__ agg, int shift4) {
  int idx = blockIdx.x * blockDim.x + threadIdx.x;
  int total = N_EDGES << shift4;
  if (idx >= total) return;
  int e = idx >> shift4;
  int c = idx & ((1 << shift4) - 1);
  int s = src[e], d = dst[e];
  float sc = onorm[s];
  const float4 v = ((const float4*)h)[((size_t)s << shift4) + c];
  float* o = agg + ((((size_t)d << shift4) + c) << 2);
  atomicAdd(o + 0, v.x * sc);
  atomicAdd(o + 1, v.y * sc);
  atomicAdd(o + 2, v.z * sc);
  atomicAdd(o + 3, v.w * sc);
}

// abf = bf16(agg * in_norm[row]);  shiftF = log2(F)
__global__ void k_convert(const float* __restrict__ agg, const float* __restrict__ inorm,
                          unsigned short* __restrict__ abf, int shiftF, int total) {
  int i = blockIdx.x * blockDim.x + threadIdx.x;
  if (i >= total) return;
  abf[i] = f32_to_bf16(agg[i] * inorm[i >> shiftF]);
}

// ---------------------------------------------------------------- WMMA GEMM
// out[M x 256] = relu(A[M x K](bf16) @ Wt^T + bias); Wt stored [256 x K] bf16.
// Block = 4 waves; each wave owns a 16x64 tile (4 accumulators share one A frag).
__global__ __launch_bounds__(128)
void k_gemm_bias_relu(const unsigned short* __restrict__ A,
                      const unsigned short* __restrict__ Wt,
                      const float* __restrict__ bias,
                      float* __restrict__ out,
                      int K) {
  const int lane   = threadIdx.x & 31;
  const int wave   = threadIdx.x >> 5;        // 0..3 -> column quadrant
  const int laneLo = lane & 15;
  const int laneHi = lane >> 4;               // 0/1
  const int row     = blockIdx.x * 16 + laneLo;
  const int colBase = wave * 64;

  const unsigned short* Arow = A + (size_t)row * K;
  const unsigned short* Wp0  = Wt + (size_t)(colBase +      laneLo) * K;
  const unsigned short* Wp1  = Wt + (size_t)(colBase + 16 + laneLo) * K;
  const unsigned short* Wp2  = Wt + (size_t)(colBase + 32 + laneLo) * K;
  const unsigned short* Wp3  = Wt + (size_t)(colBase + 48 + laneLo) * K;

  v8f acc0 = {}; v8f acc1 = {}; v8f acc2 = {}; v8f acc3 = {};

  for (int kk = 0; kk < K; kk += 32) {
    FragBF a, b0, b1, b2, b3;
    // A 16x32 bf16 as loaded here (contiguous k, same as B): lane holds row M=lane&15; elems i -> K=kA+i,
    // kA = kk + (lane>=16 ? 16 : 0)
    const int kA = kk + laneHi * 16;
    a.u[0] = *(const uint4*)(Arow + kA);
    a.u[1] = *(const uint4*)(Arow + kA + 8);
    // B 32x16 bf16 as loaded here (contiguous k, same as A): lane holds col N=lane&15; elems i -> K=kB+i,
    // kB = kk + (lane>=16 ? 16 : 0). Wt is [N][K] so 16 elems are contiguous.
    const int kB = kk + laneHi * 16;
    b0.u[0] = *(const uint4*)(Wp0 + kB); b0.u[1] = *(const uint4*)(Wp0 + kB + 8);
    b1.u[0] = *(const uint4*)(Wp1 + kB); b1.u[1] = *(const uint4*)(Wp1 + kB + 8);
    b2.u[0] = *(const uint4*)(Wp2 + kB); b2.u[1] = *(const uint4*)(Wp2 + kB + 8);
    b3.u[0] = *(const uint4*)(Wp3 + kB); b3.u[1] = *(const uint4*)(Wp3 + kB + 8);

    acc0 = __builtin_amdgcn_wmma_f32_16x16x32_bf16(false, a.v, false, b0.v, (short)0, acc0, false, false);
    acc1 = __builtin_amdgcn_wmma_f32_16x16x32_bf16(false, a.v, false, b1.v, (short)0, acc1, false, false);
    acc2 = __builtin_amdgcn_wmma_f32_16x16x32_bf16(false, a.v, false, b2.v, (short)0, acc2, false, false);
    acc3 = __builtin_amdgcn_wmma_f32_16x16x32_bf16(false, a.v, false, b3.v, (short)0, acc3, false, false);
  }

  // C/D layout: lane 0-15 -> N=lane, VGPR r -> M = r (+8 for lanes 16-31)
  const int mBase = blockIdx.x * 16 + laneHi * 8;
  v8f accs[4] = {acc0, acc1, acc2, acc3};
  #pragma unroll
  for (int j = 0; j < 4; ++j) {
    const int col = colBase + j * 16 + laneLo;
    const float bb = bias[col];
    #pragma unroll
    for (int r = 0; r < 8; ++r) {
      float v = accs[j][r] + bb;
      out[(size_t)(mBase + r) * HIDDEN + col] = fmaxf(v, 0.0f);
    }
  }
}

// ---------------------------------------------------------------- pooling + head
__global__ void k_pool(const float* __restrict__ g, const int* __restrict__ gid,
                       float* __restrict__ pool, int off) {
  int i = blockIdx.x * blockDim.x + threadIdx.x;
  if (i >= N_NODES * HIDDEN) return;
  int row = i >> 8;       // / 256
  int f   = i & 255;
  atomicAdd(&pool[gid[row] * (3 * HIDDEN) + off + f], g[i]);
}

__global__ void k_fc1(const float* __restrict__ pool, const float* __restrict__ counts,
                      const float* __restrict__ w, const float* __restrict__ b,
                      float* __restrict__ out1) {
  int g = blockIdx.x, j = threadIdx.x;      // 128 x 128
  float ic = 1.0f / fmaxf(counts[g], 1.0f);
  float s = b[j];
  const float* pg = pool + g * (3 * HIDDEN);
  for (int k = 0; k < 3 * HIDDEN; ++k) s = fmaf(pg[k] * ic, w[k * 128 + j], s);
  out1[g * 128 + j] = fmaxf(s, 0.0f);
}

__global__ void k_fc2(const float* __restrict__ h, const float* __restrict__ w,
                      const float* __restrict__ b, float* __restrict__ out) {
  int t = blockIdx.x * blockDim.x + threadIdx.x;
  if (t >= 2 * N_GRAPHS) return;
  int g = t >> 1, j = t & 1;
  float s = b[j];
  const float* hg = h + g * 128;
  for (int k = 0; k < 128; ++k) s = fmaf(hg[k], w[k * 2 + j], s);
  out[t] = 1.0f / (1.0f + __expf(-s));
}

// ---------------------------------------------------------------- launch
extern "C" void kernel_launch(void* const* d_in, const int* in_sizes, int n_in,
                              void* d_out, int out_size, void* d_ws, size_t ws_size,
                              hipStream_t stream) {
  const float* x    = (const float*)d_in[0];
  const float* W1   = (const float*)d_in[1];
  const float* b1   = (const float*)d_in[2];
  const float* W2   = (const float*)d_in[3];
  const float* b2   = (const float*)d_in[4];
  const float* W3   = (const float*)d_in[5];
  const float* b3   = (const float*)d_in[6];
  const float* fc1w = (const float*)d_in[7];
  const float* fc1b = (const float*)d_in[8];
  const float* fc2w = (const float*)d_in[9];
  const float* fc2b = (const float*)d_in[10];
  const int*   src  = (const int*)d_in[11];
  const int*   dst  = (const int*)d_in[12];
  const int*   gid  = (const int*)d_in[13];
  float* out = (float*)d_out;
  (void)in_sizes; (void)n_in; (void)out_size; (void)ws_size;

  char* ws = (char*)d_ws;
  size_t cur = 0;
  auto alloc = [&](size_t bytes) {
    void* p = ws + cur;
    cur = (cur + bytes + 255) & ~(size_t)255;
    return p;
  };
  float*          gA    = (float*)alloc((size_t)N_NODES * HIDDEN * 4);   // agg buffer
  float*          gB    = (float*)alloc((size_t)N_NODES * HIDDEN * 4);   // layer output
  unsigned short* Abf   = (unsigned short*)alloc((size_t)N_NODES * HIDDEN * 2);
  unsigned short* Wt1   = (unsigned short*)alloc((size_t)IN_FEATS * HIDDEN * 2);
  unsigned short* Wt2   = (unsigned short*)alloc((size_t)HIDDEN * HIDDEN * 2);
  unsigned short* Wt3   = (unsigned short*)alloc((size_t)HIDDEN * HIDDEN * 2);
  float*          onorm = (float*)alloc((size_t)N_NODES * 4);
  float*          inorm = (float*)alloc((size_t)N_NODES * 4);
  float*          cnts  = (float*)alloc((size_t)N_GRAPHS * 4);
  float*          pool  = (float*)alloc((size_t)N_GRAPHS * 3 * HIDDEN * 4);
  float*          fcb   = (float*)alloc((size_t)N_GRAPHS * 128 * 4);

  const int T = 256;
  auto cdiv = [](long long a, long long b) { return (unsigned)((a + b - 1) / b); };

  // degrees / counts / norms
  k_zero<<<cdiv(N_NODES, T), T, 0, stream>>>(onorm, N_NODES);
  k_zero<<<cdiv(N_NODES, T), T, 0, stream>>>(inorm, N_NODES);
  k_zero<<<1, 128, 0, stream>>>(cnts, N_GRAPHS);
  k_zero<<<cdiv(N_GRAPHS * 3 * HIDDEN, T), T, 0, stream>>>(pool, N_GRAPHS * 3 * HIDDEN);
  k_deg<<<cdiv(N_EDGES, T), T, 0, stream>>>(src, dst, onorm, inorm);
  k_counts<<<cdiv(N_NODES, T), T, 0, stream>>>(gid, cnts);
  k_rsqrt<<<cdiv(N_NODES, T), T, 0, stream>>>(onorm, N_NODES);
  k_rsqrt<<<cdiv(N_NODES, T), T, 0, stream>>>(inorm, N_NODES);

  // weights -> bf16 transposed
  k_wt_bf16<<<cdiv(IN_FEATS * HIDDEN, T), T, 0, stream>>>(W1, Wt1, IN_FEATS, HIDDEN);
  k_wt_bf16<<<cdiv(HIDDEN * HIDDEN, T), T, 0, stream>>>(W2, Wt2, HIDDEN, HIDDEN);
  k_wt_bf16<<<cdiv(HIDDEN * HIDDEN, T), T, 0, stream>>>(W3, Wt3, HIDDEN, HIDDEN);

  // ---- layer 1 (F_in = 128)
  k_zero<<<cdiv((long long)N_NODES * IN_FEATS, T), T, 0, stream>>>(gA, N_NODES * IN_FEATS);
  k_scatter<<<cdiv((long long)N_EDGES * (IN_FEATS / 4), T), T, 0, stream>>>(x, src, dst, onorm, gA, 5);
  k_convert<<<cdiv((long long)N_NODES * IN_FEATS, T), T, 0, stream>>>(gA, inorm, Abf, 7, N_NODES * IN_FEATS);
  k_gemm_bias_relu<<<N_NODES / 16, 128, 0, stream>>>(Abf, Wt1, b1, gB, IN_FEATS);
  k_pool<<<cdiv((long long)N_NODES * HIDDEN, T), T, 0, stream>>>(gB, gid, pool, 0);

  // ---- layer 2 (F_in = 256)
  k_zero<<<cdiv((long long)N_NODES * HIDDEN, T), T, 0, stream>>>(gA, N_NODES * HIDDEN);
  k_scatter<<<cdiv((long long)N_EDGES * (HIDDEN / 4), T), T, 0, stream>>>(gB, src, dst, onorm, gA, 6);
  k_convert<<<cdiv((long long)N_NODES * HIDDEN, T), T, 0, stream>>>(gA, inorm, Abf, 8, N_NODES * HIDDEN);
  k_gemm_bias_relu<<<N_NODES / 16, 128, 0, stream>>>(Abf, Wt2, b2, gB, HIDDEN);
  k_pool<<<cdiv((long long)N_NODES * HIDDEN, T), T, 0, stream>>>(gB, gid, pool, HIDDEN);

  // ---- layer 3
  k_zero<<<cdiv((long long)N_NODES * HIDDEN, T), T, 0, stream>>>(gA, N_NODES * HIDDEN);
  k_scatter<<<cdiv((long long)N_EDGES * (HIDDEN / 4), T), T, 0, stream>>>(gB, src, dst, onorm, gA, 6);
  k_convert<<<cdiv((long long)N_NODES * HIDDEN, T), T, 0, stream>>>(gA, inorm, Abf, 8, N_NODES * HIDDEN);
  k_gemm_bias_relu<<<N_NODES / 16, 128, 0, stream>>>(Abf, Wt3, b3, gB, HIDDEN);
  k_pool<<<cdiv((long long)N_NODES * HIDDEN, T), T, 0, stream>>>(gB, gid, pool, 2 * HIDDEN);

  // ---- head
  k_fc1<<<N_GRAPHS, 128, 0, stream>>>(pool, cnts, fc1w, fc1b, fcb);
  k_fc2<<<1, 2 * N_GRAPHS, 0, stream>>>(fcb, fc2w, fc2b, out);
}
